// KRadarSparseEncoderV2_34720515621007
// MI455X (gfx1250) — compile-verified, matrix operand order checked by reading, not run on hardware
//
#include <hip/hip_runtime.h>
#include <hip/hip_bf16.h>

// ---------------------------------------------------------------------------
// KRadar sparse encoder for MI455X (gfx1250, wave32, WMMA + TDM).
// All convs run as implicit GEMMs on v_wmma_f32_16x16x32_f16.
// Weight tiles are staged into LDS by the Tensor Data Mover (tensor_load_to_lds)
// from pre-transposed [tap][co][ci] weights so fragments read contiguously;
// activations gather through global_load_b128 + ds_store_b128.
// Activations: fp16 NDHWC. Accumulation: f32. Final output: f32 NCHW (2,768,160,160).
// ---------------------------------------------------------------------------

typedef __attribute__((ext_vector_type(16))) _Float16 v16h;
typedef __attribute__((ext_vector_type(8)))  _Float16 v8h;
typedef __attribute__((ext_vector_type(8)))  float    v8f;

#define TM 128   // output rows (spatial positions) per workgroup tile
#define TN 64    // output channels per workgroup tile
#define TKK 32   // K step (f16 WMMA K = 32)

#if __has_builtin(__builtin_amdgcn_tensor_load_to_lds)
#define USE_TDM 1
typedef unsigned int u32x4 __attribute__((ext_vector_type(4)));
typedef int          i32x8 __attribute__((ext_vector_type(8)));
typedef int          i32x4 __attribute__((ext_vector_type(4)));
#endif

struct ConvDims {
  int Di, Hi, Wi, Ci;     // input dims (N fixed = 2), NDHWC
  int Do, Ho, Wo, Co;     // output dims
  int kd, kh, kw;         // kernel
  int stride, pad_d, pad_hw;
};

static __device__ __forceinline__ v8f wmma32(v16h a, v16h b, v8f c) {
  // 8 args: (neg_a, A, neg_b, B, c_mod, C, reuse_a, reuse_b)
  return __builtin_amdgcn_wmma_f32_16x16x32_f16(false, a, false, b, (short)0, c, false, false);
}

// Fragment load from an LDS tile stored row-major [row][k], stride ld.
// Lane striping per ISA 7.12.2 (16-bit matrices, K=32):
//   lanes 0-15 : row = lane,    K chunks {0..7, 16..23}
//   lanes 16-31: row = lane-16, K chunks {8..15, 24..31}
// Used for both A (row = M) and B (row = N; weights pre-transposed to [co][ci]).
static __device__ __forceinline__ v16h load_frag(const _Float16* base, int ld, int lane) {
  const int r  = lane & 15;
  const int k0 = (lane & 16) ? 8 : 0;
  const _Float16* p = base + r * ld + k0;
  v16h v;
#pragma unroll
  for (int i = 0; i < 8; ++i) v[i] = p[i];
#pragma unroll
  for (int i = 0; i < 8; ++i) v[8 + i] = p[16 + i];
  return v;
}

#ifdef USE_TDM
// Issue one TDM 2D tile load: TN rows x TKK cols of f16 weights into LDS.
// gsrc points at element (row0, col0) of a [rows][Ci] tensor with row pitch
// stride_elems; width_rem = cols remaining from col0 (OOB clamp). ISA ch.8.
static __device__ __forceinline__ void tdm_load_b_tile(const _Float16* gsrc, unsigned lds_off,
                                                       int width_rem, int stride_elems) {
  const unsigned long long ga = (unsigned long long)(const void*)gsrc;
  u32x4 g0;
  g0[0] = 1u;                                               // count=1, user descriptor
  g0[1] = lds_off;                                          // lds_addr (bytes)
  g0[2] = (unsigned)(ga & 0xFFFFFFFFull);                   // global_addr[31:0]
  g0[3] = (unsigned)((ga >> 32) & 0x01FFFFFFull) | (2u << 30); // addr[56:32] | type=2
  i32x8 g1;
  g1[0] = (1 << 16);                                        // data_size = 2 bytes
  g1[1] = (int)((unsigned)(width_rem & 0xFFFF) << 16);      // tensor_dim0[15:0]
  g1[2] = (int)(((unsigned)width_rem >> 16) & 0xFFFFu) | (TN << 16); // dim0 hi | dim1 lo
  g1[3] = (int)(TKK << 16);                                 // dim1 hi(0) | tile_dim0 = TKK
  g1[4] = TN;                                               // tile_dim1 = TN, tile_dim2 = 0
  g1[5] = stride_elems;                                     // tensor_dim0_stride[31:0]
  g1[6] = 0;                                                // stride0 hi | stride1 lo
  g1[7] = 0;
  i32x4 z4; z4[0] = 0; z4[1] = 0; z4[2] = 0; z4[3] = 0;
  i32x8 z8; z8[0] = 0; z8[1] = 0; z8[2] = 0; z8[3] = 0;
  z8[4] = 0; z8[5] = 0; z8[6] = 0; z8[7] = 0;
  // 6-arg form (clang-23 / therock-10.0): groups 0..3 + extra group + cpol
  __builtin_amdgcn_tensor_load_to_lds(g0, g1, z4, z4, z8, 0);
}
#endif

// ---------------------------------------------------------------------------
// Implicit-GEMM 3D conv + folded BN + ReLU + spatial mask, fp16 in/out.
// in: NDHWC f16; wt: [kd*kh*kw][Co][Ci] f16 (transposed); out: NDHWC f16.
// ---------------------------------------------------------------------------
__global__ __launch_bounds__(256)
void conv3d_bn_wmma(const _Float16* __restrict__ in, _Float16* __restrict__ out,
                    const _Float16* __restrict__ wt,
                    const float* __restrict__ alpha, const float* __restrict__ beta,
                    const _Float16* __restrict__ mask,
                    ConvDims cd, int relu) {
  __shared__ _Float16 lA[TM][TKK];   // 8 KB
  __shared__ _Float16 lB[TN][TKK];   // 4 KB, row-major [n][k]

  const int tid  = threadIdx.x;
  const int lane = tid & 31;
  const int wave = tid >> 5;
  const int Mtot = 2 * cd.Do * cd.Ho * cd.Wo;
  const int m0 = blockIdx.x * TM;
  const int n0 = blockIdx.y * TN;
  const int wm = (wave & 3) * 32;   // wave M offset inside tile
  const int wn = (wave >> 2) * 32;  // wave N offset inside tile

  const v8f vz = {0.f, 0.f, 0.f, 0.f, 0.f, 0.f, 0.f, 0.f};
  v8f acc[2][2];
  acc[0][0] = vz; acc[0][1] = vz; acc[1][0] = vz; acc[1][1] = vz;

  // A tile: each thread owns one output row and half of its 32 channels.
  const int rowA  = tid >> 1;
  const int chunk = (tid & 1) * 16;
  const int gmA   = m0 + rowA;
  const bool mval = gmA < Mtot;
  int tt = mval ? gmA : 0;
  const int ow = tt % cd.Wo; tt /= cd.Wo;
  const int oh = tt % cd.Ho; tt /= cd.Ho;
  const int od = tt % cd.Do;
  const int nb = tt / cd.Do;
  const int id0 = od * cd.stride - cd.pad_d;
  const int ih0 = oh * cd.stride - cd.pad_hw;
  const int iw0 = ow * cd.stride - cd.pad_hw;
#ifdef USE_TDM
  const unsigned ldsB = (unsigned)(unsigned long long)(void*)&lB[0][0];
#else
  const int brow = tid >> 2;        // co row 0..63
  const int bcol = (tid & 3) * 8;   // ci chunk
#endif

  const int khw  = cd.kh * cd.kw;
  const int taps = cd.kd * khw;

  for (int tap = 0; tap < taps; ++tap) {
    const int tkd  = tap / khw;
    const int trem = tap - tkd * khw;
    const int tkh  = trem / cd.kw;
    const int tkw  = trem - tkh * cd.kw;
    const int id = id0 + tkd, ih = ih0 + tkh, iw = iw0 + tkw;
    const bool inb = mval &&
                     (unsigned)id < (unsigned)cd.Di &&
                     (unsigned)ih < (unsigned)cd.Hi &&
                     (unsigned)iw < (unsigned)cd.Wi;
    const _Float16* srcp =
        in + ((((long)nb * cd.Di + id) * cd.Hi + ih) * (long)cd.Wi + iw) * cd.Ci;
    const _Float16* wtap = wt + (long)tap * cd.Ci * cd.Co;   // [Co][Ci] slice

    for (int kc = 0; kc < cd.Ci; kc += TKK) {
      v8h a0 = {(_Float16)0,(_Float16)0,(_Float16)0,(_Float16)0,
                (_Float16)0,(_Float16)0,(_Float16)0,(_Float16)0};
      v8h a1 = a0;
      if (inb) {
        a0 = *(const v8h*)(srcp + kc + chunk);
        a1 = *(const v8h*)(srcp + kc + chunk + 8);
        __builtin_prefetch(srcp + kc + TKK + chunk, 0, 1);  // global_prefetch_b8
      }
#ifndef USE_TDM
      const v8h wv = *(const v8h*)(wtap + (long)(n0 + brow) * cd.Ci + kc + bcol);
#endif

      __syncthreads();  // previous iteration's fragment reads done
#ifdef USE_TDM
      if (wave == 0)    // TDM ignores EXEC -> gate at wave granularity
        tdm_load_b_tile(wtap + (long)n0 * cd.Ci + kc, ldsB, cd.Ci - kc, cd.Ci);
#endif
      *(v8h*)&lA[rowA][chunk]     = a0;
      *(v8h*)&lA[rowA][chunk + 8] = a1;
#ifdef USE_TDM
      if (wave == 0) __builtin_amdgcn_s_wait_tensorcnt(0);
#else
      *(v8h*)&lB[brow][bcol] = wv;
#endif
      __syncthreads();  // tiles ready (incl. TDM completion observed by wave 0)

      const v16h fa0 = load_frag(&lA[wm][0],      TKK, lane);
      const v16h fa1 = load_frag(&lA[wm + 16][0], TKK, lane);
      const v16h fb0 = load_frag(&lB[wn][0],      TKK, lane);
      const v16h fb1 = load_frag(&lB[wn + 16][0], TKK, lane);
      acc[0][0] = wmma32(fa0, fb0, acc[0][0]);
      acc[0][1] = wmma32(fa0, fb1, acc[0][1]);
      acc[1][0] = wmma32(fa1, fb0, acc[1][0]);
      acc[1][1] = wmma32(fa1, fb1, acc[1][1]);
    }
  }

  // ---- epilogue: y = relu(acc*alpha + beta) * mask, store NDHWC fp16 -------
  const int mhalf = (lane < 16) ? 0 : 8;
  const int ncol  = lane & 15;
#pragma unroll
  for (int mi = 0; mi < 2; ++mi) {
#pragma unroll
    for (int ni = 0; ni < 2; ++ni) {
      const int n  = n0 + wn + ni * 16 + ncol;
      const float al = alpha[n];
      const float be = beta[n];
#pragma unroll
      for (int r = 0; r < 8; ++r) {
        const int m = m0 + wm + mi * 16 + mhalf + r;
        if (m < Mtot) {
          float v = acc[mi][ni][r] * al + be;
          if (relu) v = v > 0.f ? v : 0.f;
          if (mask) v *= (float)mask[m];
          out[(long)m * cd.Co + n] = (_Float16)v;
        }
      }
    }
  }
}

// ---------------------------------------------------------------------------
// Transposed 2D conv (lhs-dilated gather form) + folded BN, f32 NCHW output.
// For stride s, each output phase sees exactly (k/s)^2 == 9 real taps.
// bev: NHWC f16 (2,Hb,Hb,Ci); wt: [kh][kw][256][Ci] f16 (pre-flipped+transposed);
// out: (2,768,160,160) f32, channels [coff, coff+256).
// ---------------------------------------------------------------------------
__global__ __launch_bounds__(256)
void trconv_bn_wmma(const _Float16* __restrict__ bev, float* __restrict__ out,
                    const _Float16* __restrict__ wt,
                    const float* __restrict__ alpha, const float* __restrict__ beta,
                    int Hb, int Ci, int s, int k, int pad, int coff) {
  __shared__ _Float16 lA[TM][TKK];
  __shared__ _Float16 lB[TN][TKK];

  const int tid  = threadIdx.x;
  const int lane = tid & 31;
  const int wave = tid >> 5;
  const int Co = 256;

  const int phase = blockIdx.z;
  const int py = phase / s, px = phase % s;
  const int offy = (py + pad) / s, ry = (py + pad) % s;
  const int offx = (px + pad) / s, rx = (px + pad) % s;
  const int Hq = 160 / s;
  const int Mtot = 2 * Hq * Hq;
  const int m0 = blockIdx.x * TM;
  const int n0 = blockIdx.y * TN;
  const int wm = (wave & 3) * 32;
  const int wn = (wave >> 2) * 32;

  const v8f vz = {0.f, 0.f, 0.f, 0.f, 0.f, 0.f, 0.f, 0.f};
  v8f acc[2][2];
  acc[0][0] = vz; acc[0][1] = vz; acc[1][0] = vz; acc[1][1] = vz;

  const int rowA  = tid >> 1;
  const int chunk = (tid & 1) * 16;
  int gmA = m0 + rowA; if (gmA >= Mtot) gmA = Mtot - 1;
  int tt = gmA;
  const int qx = tt % Hq; tt /= Hq;
  const int qy = tt % Hq;
  const int nb = tt / Hq;
  const int ihb = qy + offy, iwb = qx + offx;
#ifdef USE_TDM
  const unsigned ldsB = (unsigned)(unsigned long long)(void*)&lB[0][0];
#else
  const int brow = tid >> 2;
  const int bcol = (tid & 3) * 8;
#endif

  for (int tap = 0; tap < 9; ++tap) {
    const int th = tap / 3, tw = tap % 3;
    const int ih = ihb - th, iw = iwb - tw;
    const bool inb = (unsigned)ih < (unsigned)Hb && (unsigned)iw < (unsigned)Hb;
    const _Float16* srcp = bev + (((long)nb * Hb + ih) * Hb + iw) * Ci;
    const int kh = ry + s * th;
    const int kw = rx + s * tw;
    const _Float16* wtap = wt + ((long)kh * k + kw) * Ci * Co;  // [Co][Ci] slice

    for (int kc = 0; kc < Ci; kc += TKK) {
      v8h a0 = {(_Float16)0,(_Float16)0,(_Float16)0,(_Float16)0,
                (_Float16)0,(_Float16)0,(_Float16)0,(_Float16)0};
      v8h a1 = a0;
      if (inb) {
        a0 = *(const v8h*)(srcp + kc + chunk);
        a1 = *(const v8h*)(srcp + kc + chunk + 8);
        __builtin_prefetch(srcp + kc + TKK + chunk, 0, 1);
      }
#ifndef USE_TDM
      const v8h wv = *(const v8h*)(wtap + (long)(n0 + brow) * Ci + kc + bcol);
#endif

      __syncthreads();
#ifdef USE_TDM
      if (wave == 0)
        tdm_load_b_tile(wtap + (long)n0 * Ci + kc, ldsB, Ci - kc, Ci);
#endif
      *(v8h*)&lA[rowA][chunk]     = a0;
      *(v8h*)&lA[rowA][chunk + 8] = a1;
#ifdef USE_TDM
      if (wave == 0) __builtin_amdgcn_s_wait_tensorcnt(0);
#else
      *(v8h*)&lB[brow][bcol] = wv;
#endif
      __syncthreads();

      const v16h fa0 = load_frag(&lA[wm][0],      TKK, lane);
      const v16h fa1 = load_frag(&lA[wm + 16][0], TKK, lane);
      const v16h fb0 = load_frag(&lB[wn][0],      TKK, lane);
      const v16h fb1 = load_frag(&lB[wn + 16][0], TKK, lane);
      acc[0][0] = wmma32(fa0, fb0, acc[0][0]);
      acc[0][1] = wmma32(fa0, fb1, acc[0][1]);
      acc[1][0] = wmma32(fa1, fb0, acc[1][0]);
      acc[1][1] = wmma32(fa1, fb1, acc[1][1]);
    }
  }

  const int mhalf = (lane < 16) ? 0 : 8;
  const int ncol  = lane & 15;
#pragma unroll
  for (int mi = 0; mi < 2; ++mi) {
#pragma unroll
    for (int ni = 0; ni < 2; ++ni) {
      const int n  = n0 + wn + ni * 16 + ncol;
      const float al = alpha[n];
      const float be = beta[n];
#pragma unroll
      for (int r = 0; r < 8; ++r) {
        const int m = m0 + wm + mi * 16 + mhalf + r;
        if (m < Mtot) {
          int t2 = m;
          const int qx2 = t2 % Hq; t2 /= Hq;
          const int qy2 = t2 % Hq;
          const int nb2 = t2 / Hq;
          const int y = s * qy2 + py;
          const int x = s * qx2 + px;
          out[(((long)nb2 * 768 + coff + n) * 160 + y) * 160 + x] = acc[mi][ni][r] * al + be;
        }
      }
    }
  }
}

// ---------------------------------------------------------------------------
// Small helper kernels
// ---------------------------------------------------------------------------
__global__ void zero_h(_Float16* p, long n8) {
  const long i = (long)blockIdx.x * blockDim.x + threadIdx.x;
  if (i < n8) {
    const v8h z = {(_Float16)0,(_Float16)0,(_Float16)0,(_Float16)0,
                   (_Float16)0,(_Float16)0,(_Float16)0,(_Float16)0};
    ((v8h*)p)[i] = z;
  }
}

// Fused scatter + 1x1x1 input conv: x0[b,z,y,x,c] = W_in[c,:].feat + b_in[c]; mask = 1.
__global__ void scatter_in(const float* __restrict__ feat, const int* __restrict__ coors,
                           const float* __restrict__ w_in, const float* __restrict__ b_in,
                           _Float16* __restrict__ x0, _Float16* __restrict__ mask, int nvox) {
  const long t = (long)blockIdx.x * blockDim.x + threadIdx.x;
  const int vox = (int)(t >> 6);
  const int c   = (int)(t & 63);
  if (vox >= nvox) return;
  const int b = coors[vox * 4 + 0];
  const int z = coors[vox * 4 + 1];
  const int y = coors[vox * 4 + 2];
  const int x = coors[vox * 4 + 3];
  const float f0 = feat[vox * 4 + 0], f1 = feat[vox * 4 + 1];
  const float f2 = feat[vox * 4 + 2], f3 = feat[vox * 4 + 3];
  const float v = w_in[c * 4 + 0] * f0 + w_in[c * 4 + 1] * f1 +
                  w_in[c * 4 + 2] * f2 + w_in[c * 4 + 3] * f3 + b_in[c];
  const long p = (((long)b * 8 + z) * 160 + y) * 160 + x;
  x0[p * 64 + c] = (_Float16)v;
  if (c == 0) mask[p] = (_Float16)1.0f;
}

// Binary dilation of the occupancy mask (kh==kw==khw square kernel).
__global__ void mask_dilate(const _Float16* __restrict__ src, _Float16* __restrict__ dst,
                            int Di, int Hi, int Wi, int Do, int Ho, int Wo,
                            int kd, int khw, int stride, int pad_d, int pad_hw) {
  const int t = blockIdx.x * blockDim.x + threadIdx.x;
  const int tot = 2 * Do * Ho * Wo;
  if (t >= tot) return;
  int tt = t;
  const int ow = tt % Wo; tt /= Wo;
  const int oh = tt % Ho; tt /= Ho;
  const int od = tt % Do;
  const int nb = tt / Do;
  const int id0 = od * stride - pad_d;
  const int ih0 = oh * stride - pad_hw;
  const int iw0 = ow * stride - pad_hw;
  float r = 0.f;
  for (int a = 0; a < kd; ++a) {
    const int id = id0 + a; if ((unsigned)id >= (unsigned)Di) continue;
    for (int b = 0; b < khw; ++b) {
      const int ih = ih0 + b; if ((unsigned)ih >= (unsigned)Hi) continue;
      for (int cc = 0; cc < khw; ++cc) {
        const int iw = iw0 + cc; if ((unsigned)iw >= (unsigned)Wi) continue;
        if ((float)src[(((long)nb * Di + id) * Hi + ih) * (long)Wi + iw] > 0.f) r = 1.f;
      }
    }
  }
  dst[t] = (_Float16)r;
}

// f32 [co][ci][kd][kh][kw] -> f16 [kd][kh][kw][co][ci]  (transposed for TDM tiles)
__global__ void pack_w3d(const float* __restrict__ src, _Float16* __restrict__ dst,
                         int Ci, int Co, int KD, int KH, int KW) {
  const long t = (long)blockIdx.x * blockDim.x + threadIdx.x;
  const long tot = (long)Ci * Co * KD * KH * KW;
  if (t >= tot) return;
  const int i = (int)(t % Ci); long r = t / Ci;
  const int o = (int)(r % Co); long tap = r / Co;
  const int kw = (int)(tap % KW); long r2 = tap / KW;
  const int kh = (int)(r2 % KH);
  const int kd = (int)(r2 / KH);
  dst[t] = (_Float16)src[((((long)o * Ci + i) * KD + kd) * KH + kh) * KW + kw];
}

// w_tr f32 [ci][co][k][k] -> f16 [kh][kw][co][ci] with spatial flip
// (flip+transpose of the reference's rhs construction, plus co/ci transpose).
__global__ void pack_wtr(const float* __restrict__ src, _Float16* __restrict__ dst,
                         int Ci, int Co, int K) {
  const long t = (long)blockIdx.x * blockDim.x + threadIdx.x;
  const long tot = (long)Ci * Co * K * K;
  if (t >= tot) return;
  const int i = (int)(t % Ci); long r = t / Ci;
  const int o = (int)(r % Co); long tap = r / Co;
  const int kw = (int)(tap % K);
  const int kh = (int)(tap / K);
  dst[t] = (_Float16)src[(((long)i * Co + o) * K + (K - 1 - kh)) * K + (K - 1 - kw)];
}

// Fold conv-bias + BN into per-channel alpha/beta: y = x*alpha + beta.
__global__ void bn_fold(const float* __restrict__ bconv, const float* __restrict__ g,
                        const float* __restrict__ bb, const float* __restrict__ m,
                        const float* __restrict__ v, float* __restrict__ alpha,
                        float* __restrict__ beta, int C) {
  const int c = blockIdx.x * blockDim.x + threadIdx.x;
  if (c < C) {
    const float a = g[c] * rsqrtf(v[c] + 1e-5f);
    alpha[c] = a;
    beta[c]  = (bconv[c] - m[c]) * a + bb[c];
  }
}

// ---------------------------------------------------------------------------
// Host orchestration
// ---------------------------------------------------------------------------
extern "C" void kernel_launch(void* const* d_in, const int* in_sizes, int n_in,
                              void* d_out, int out_size, void* d_ws, size_t ws_size,
                              hipStream_t stream) {
  (void)n_in; (void)out_size; (void)ws_size;
  // Input index map (params flattened in insertion order):
  // 0: voxel_features, 1: coors, 2: batch_size, 3: w_in, 4: b_in,
  // then per stage (base = 5 + 30*si):
  //   +0 w_sp +1 b_sp +2..5 bn_sp(g,b,m,v) +6 w_sa +7 b_sa +8..11 bn_sa
  //   +12 w_sb +13 b_sb +14..17 bn_sb +18 w_bev +19 b_bev +20..23 bn_bev
  //   +24 w_tr +25 b_tr +26..29 bn_tr
  auto FP = [&](int i) { return (const float*)d_in[i]; };

  // ---- workspace layout ----------------------------------------------------
  char* ws = (char*)d_ws;
  const size_t SZ_BUF = (size_t)2 * 8 * 160 * 160 * 64 * sizeof(_Float16);  // 52.4 MB
  const size_t SZ_BEV = (size_t)2 * 160 * 160 * 64 * sizeof(_Float16);      // 6.55 MB
  _Float16* bufA  = (_Float16*)ws;
  _Float16* bufB  = (_Float16*)(ws + SZ_BUF);
  _Float16* bevb  = (_Float16*)(ws + 2 * SZ_BUF);
  size_t off = 2 * SZ_BUF + SZ_BEV;
  _Float16* maskA = (_Float16*)(ws + off); off += (1u << 20);
  _Float16* maskB = (_Float16*)(ws + off); off += (1u << 20);
  _Float16* maskV = (_Float16*)(ws + off); off += (1u << 20);
  _Float16* wbase = (_Float16*)(ws + off);
  float*    abf   = (float*)(ws + off + (size_t)40 * 1024 * 1024);

  const int CINs[3] = {64, 64, 128};
  const int CBs[3]  = {64, 128, 256};
  const int ZKs[3]  = {8, 4, 2};
  const int STs[3]  = {1, 2, 2};
  const int TKs[3]  = {3, 6, 12};
  const int TSs[3]  = {1, 2, 4};
  const int TPs[3]  = {1, 2, 4};

  // ---- weight prepack + BN folding ----------------------------------------
  _Float16* wp[3][5];
  float* alp[3][5];
  float* bep[3][5];
  size_t wofs = 0, abofs = 0;
  for (int si = 0; si < 3; ++si) {
    const int base = 5 + 30 * si;
    const int widx[5] = {base + 0, base + 6, base + 12, base + 18, base + 24};
    const int bcix[5] = {base + 1, base + 7, base + 13, base + 19, base + 25};
    const int gidx[5] = {base + 2, base + 8, base + 14, base + 20, base + 26};
    for (int cv = 0; cv < 5; ++cv) {
      const int ci = (cv == 0) ? CINs[si] : CBs[si];
      const int co = (cv == 4) ? 256 : CBs[si];
      long wel;
      if (cv < 3)       wel = (long)ci * co * 27;
      else if (cv == 3) wel = (long)ci * co * ZKs[si];
      else              wel = (long)ci * co * TKs[si] * TKs[si];
      wp[si][cv]  = wbase + wofs; wofs  += (size_t)wel;
      alp[si][cv] = abf + abofs;  abofs += co;
      bep[si][cv] = abf + abofs;  abofs += co;
      const int blocks = (int)((wel + 255) / 256);
      if (cv < 3)
        pack_w3d<<<blocks, 256, 0, stream>>>(FP(widx[cv]), wp[si][cv], ci, co, 3, 3, 3);
      else if (cv == 3)
        pack_w3d<<<blocks, 256, 0, stream>>>(FP(widx[cv]), wp[si][cv], ci, co, ZKs[si], 1, 1);
      else
        pack_wtr<<<blocks, 256, 0, stream>>>(FP(widx[cv]), wp[si][cv], ci, co, TKs[si]);
      bn_fold<<<(co + 63) / 64, 64, 0, stream>>>(FP(bcix[cv]), FP(gidx[cv]), FP(gidx[cv] + 1),
                                                 FP(gidx[cv] + 2), FP(gidx[cv] + 3),
                                                 alp[si][cv], bep[si][cv], co);
    }
  }

  // ---- scatter voxels + fused 1x1x1 input conv ----------------------------
  {
    const long n8 = (2L * 8 * 160 * 160 * 64) / 8;
    zero_h<<<(int)((n8 + 255) / 256), 256, 0, stream>>>(bufA, n8);
    const long m8 = (2L * 8 * 160 * 160) / 8;
    zero_h<<<(int)((m8 + 255) / 256), 256, 0, stream>>>(maskA, m8);
    const int nvox = in_sizes[0] / 4;
    const long sth = (long)nvox * 64;
    scatter_in<<<(int)((sth + 255) / 256), 256, 0, stream>>>(
        (const float*)d_in[0], (const int*)d_in[1], FP(3), FP(4), bufA, maskA, nvox);
  }

  // ---- three encoder stages ------------------------------------------------
  _Float16* curX = bufA; _Float16* altX = bufB;
  _Float16* curM = maskA; _Float16* altM = maskB;
  int cin = 64, Di = 8, Hi = 160;
  for (int si = 0; si < 3; ++si) {
    const int c = CBs[si], st = STs[si];
    const int Dn = (Di + 2 - 3) / st + 1;
    const int Hn = (Hi + 2 - 3) / st + 1;
    const int Mo = 2 * Dn * Hn * Hn;

    mask_dilate<<<(Mo + 255) / 256, 256, 0, stream>>>(curM, altM, Di, Hi, Hi, Dn, Hn, Hn,
                                                      3, 3, st, 1, 1);
    ConvDims cd0 = {Di, Hi, Hi, cin, Dn, Hn, Hn, c, 3, 3, 3, st, 1, 1};
    conv3d_bn_wmma<<<dim3((Mo + TM - 1) / TM, c / 64), 256, 0, stream>>>(
        curX, altX, wp[si][0], alp[si][0], bep[si][0], altM, cd0, 1);
    ConvDims cd1 = {Dn, Hn, Hn, c, Dn, Hn, Hn, c, 3, 3, 3, 1, 1, 1};
    conv3d_bn_wmma<<<dim3((Mo + TM - 1) / TM, c / 64), 256, 0, stream>>>(
        altX, curX, wp[si][1], alp[si][1], bep[si][1], altM, cd1, 1);
    conv3d_bn_wmma<<<dim3((Mo + TM - 1) / TM, c / 64), 256, 0, stream>>>(
        curX, altX, wp[si][2], alp[si][2], bep[si][2], altM, cd1, 1);

    const int Mb = 2 * Hn * Hn;
    mask_dilate<<<(Mb + 255) / 256, 256, 0, stream>>>(altM, maskV, Dn, Hn, Hn, 1, Hn, Hn,
                                                      ZKs[si], 1, 1, 0, 0);
    ConvDims cdb = {Dn, Hn, Hn, c, 1, Hn, Hn, c, ZKs[si], 1, 1, 1, 0, 0};
    conv3d_bn_wmma<<<dim3((Mb + TM - 1) / TM, c / 64), 256, 0, stream>>>(
        altX, bevb, wp[si][3], alp[si][3], bep[si][3], maskV, cdb, 1);

    const int sv = TSs[si], kk = TKs[si], pd = kk - 1 - TPs[si];
    const int Hq = 160 / sv;
    const int Mp = 2 * Hq * Hq;
    trconv_bn_wmma<<<dim3((Mp + TM - 1) / TM, 4, sv * sv), 256, 0, stream>>>(
        bevb, (float*)d_out, wp[si][4], alp[si][4], bep[si][4], Hn, c, sv, kk, pd, 256 * si);

    _Float16* t;
    t = curX; curX = altX; altX = t;
    t = curM; curM = altM; altM = t;
    cin = c; Di = Dn; Hi = Hn;
  }
}
